// QuantConv1DBNBlock_84585085927957
// MI455X (gfx1250) — 8-bit matrix operand order repaired and checked by reading, compiled, not run on hardware
//
#include <hip/hip_runtime.h>

typedef __attribute__((ext_vector_type(8))) int v8i;

#define T_IN    2048
#define T_OUT   2046
#define CIN     128
#define COUT    256
#define BATCH   64
#define KSTEPS  6
#define NTILES  16
#define TTILES  128
#define NPW     4
#define NGROUPS (NTILES / NPW)
#define BN_EPS  1e-3f

__global__ void __launch_bounds__(256) quant_x_kernel(
    const float* __restrict__ x, unsigned* __restrict__ xq, int n4) {
  int i = blockIdx.x * blockDim.x + threadIdx.x;
  if (i >= n4) return;
  float4 v = ((const float4*)x)[i];
  unsigned b0 = (v.x >= 0.f) ? 0x01u : 0xFFu;
  unsigned b1 = (v.y >= 0.f) ? 0x01u : 0xFFu;
  unsigned b2 = (v.z >= 0.f) ? 0x01u : 0xFFu;
  unsigned b3 = (v.w >= 0.f) ? 0x01u : 0xFFu;
  xq[i] = b0 | (b1 << 8) | (b2 << 16) | (b3 << 24);
}

__global__ void __launch_bounds__(256) build_bfrag_kernel(
    const float* __restrict__ w, unsigned* __restrict__ bf) {
  int tid = blockIdx.x * blockDim.x + threadIdx.x;
  if (tid >= NTILES * KSTEPS * 32) return;
  int lane  = tid & 31;
  int ktile = (tid >> 5) % KSTEPS;
  int ntile = tid / (32 * KSTEPS);
  int co = ntile * 16 + (lane & 15);
  int hi = lane >> 4;
  #pragma unroll
  for (int v = 0; v < 8; ++v) {
    int base = (v >> 2) * 32 + hi * 16 + (v & 3) * 4;
    unsigned d = 0;
    #pragma unroll
    for (int j = 0; j < 4; ++j) {
      int Kg = ktile * 64 + base + j;
      int k  = Kg >> 7;
      int ci = Kg & 127;
      float val = w[(k * CIN + ci) * COUT + co];
      unsigned byte = (val >= 0.f) ? 0x01u : 0xFFu;
      d |= byte << (8 * j);
    }
    bf[(((size_t)ntile * KSTEPS + ktile) * 32 + lane) * 8 + v] = d;
  }
}

__global__ void __launch_bounds__(256) bconv_wmma_kernel(
    const unsigned* __restrict__ xq,
    const unsigned* __restrict__ bf,
    const float* __restrict__ bias,
    const float* __restrict__ beta,
    const float* __restrict__ mean,
    const float* __restrict__ var,
    float* __restrict__ out) {
  const int wid  = blockIdx.x * 8 + (threadIdx.x >> 5);
  const int lane = threadIdx.x & 31;
  const int ngroup = wid & (NGROUPS - 1);
  const int ttile  = (wid >> 2) & (TTILES - 1);
  const int b      = wid >> 9;

  const int mrow = lane & 15;
  const int hi   = lane >> 4;
  int t = ttile * 16 + mrow;
  int tload = (t > T_OUT - 1) ? (T_OUT - 1) : t;

  v8i acc0 = {}, acc1 = {}, acc2 = {}, acc3 = {};

  #pragma unroll
  for (int ks = 0; ks < KSTEPS; ++ks) {
    int tt = tload + (ks >> 1);
    const unsigned* arow = xq + ((size_t)b * T_IN + tt) * (CIN / 4);
    int ci4 = (ks & 1) * 16;
    v8i a;
    #pragma unroll
    for (int v = 0; v < 8; ++v) {
      int d4 = ci4 + (v >> 2) * 8 + (v & 3) + hi * 4;
      a[v] = (int)arow[d4];
    }
    const unsigned* bbase =
        bf + (((size_t)(ngroup * NPW) * KSTEPS + ks) * 32 + lane) * 8;
    const size_t bstride = (size_t)KSTEPS * 32 * 8;
    v8i bm0 = *(const v8i*)(bbase + 0 * bstride);
    v8i bm1 = *(const v8i*)(bbase + 1 * bstride);
    v8i bm2 = *(const v8i*)(bbase + 2 * bstride);
    v8i bm3 = *(const v8i*)(bbase + 3 * bstride);

    acc0 = __builtin_amdgcn_wmma_i32_16x16x64_iu8(true, a, true, bm0, acc0, false, false);
    acc1 = __builtin_amdgcn_wmma_i32_16x16x64_iu8(true, a, true, bm1, acc1, false, false);
    acc2 = __builtin_amdgcn_wmma_i32_16x16x64_iu8(true, a, true, bm2, acc2, false, false);
    acc3 = __builtin_amdgcn_wmma_i32_16x16x64_iu8(true, a, true, bm3, acc3, false, false);
  }

  v8i accs[NPW] = {acc0, acc1, acc2, acc3};
  #pragma unroll
  for (int i = 0; i < NPW; ++i) {
    int co = (ngroup * NPW + i) * 16 + mrow;
    float inv   = __frsqrt_rn(var[co] + BN_EPS);
    float shift = (bias[co] - mean[co]) * inv + beta[co];
    #pragma unroll
    for (int r = 0; r < 8; ++r) {
      int M  = r + hi * 8;
      int to = ttile * 16 + M;
      if (to < T_OUT) {
        out[((size_t)b * T_OUT + to) * COUT + co] =
            (float)accs[i][r] * inv + shift;
      }
    }
  }
}

extern "C" void kernel_launch(void* const* d_in, const int* in_sizes, int n_in,
                              void* d_out, int out_size, void* d_ws, size_t ws_size,
                              hipStream_t stream) {
  const float* x      = (const float*)d_in[0];
  const float* kernel = (const float*)d_in[1];
  const float* bias   = (const float*)d_in[2];
  const float* beta   = (const float*)d_in[3];
  const float* mmean  = (const float*)d_in[4];
  const float* mvar   = (const float*)d_in[5];
  float* out = (float*)d_out;

  unsigned* xq = (unsigned*)d_ws;
  unsigned* bf = (unsigned*)((char*)d_ws + (size_t)BATCH * T_IN * CIN);

  {
    int n4 = BATCH * T_IN * CIN / 4;
    quant_x_kernel<<<n4 / 256, 256, 0, stream>>>(x, xq, n4);
  }
  {
    int nthr = NTILES * KSTEPS * 32;
    build_bfrag_kernel<<<(nthr + 255) / 256, 256, 0, stream>>>(kernel, bf);
  }
  {
    int nblocks = BATCH * TTILES * NGROUPS / 8;
    bconv_wmma_kernel<<<nblocks, 256, 0, stream>>>(xq, bf, bias, beta,
                                                   mmean, mvar, out);
  }
}
